// AttenLSTMModel_76450417869161
// MI455X (gfx1250) — matrix operand order repaired and checked by reading, not compiled, not run on hardware
//
#include <hip/hip_runtime.h>

typedef __bf16 bf16_t;
typedef __attribute__((ext_vector_type(16))) __bf16 v16bf;
typedef __attribute__((ext_vector_type(8)))  float  v8f;

#define LSTM_B 64
#define LSTM_T 512
#define LSTM_I 128
#define LSTM_H 512
#define LSTM_G 2048   // 4*H
#define LSTM_O 128

// ---------------------------------------------------------------------------
// WMMA fragment helpers (CDNA5 16x16x32 bf16, wave32)
// A-matrix 16x32 (MxK): lanes 0-15 -> M=lane, K in {0..7,16..23};
//                       lanes 16-31 -> M=lane-16, K in {8..15,24..31}
// ---------------------------------------------------------------------------
__device__ __forceinline__ v16bf load_a_frag(const bf16_t* A, int lda) {
  const int lane = threadIdx.x & 31;
  const int m  = lane & 15;
  const int kb = (lane >> 4) << 3;         // 0 or 8
  const bf16_t* p = A + m * lda + kb;
  union { v16bf v; uint4 q[2]; } u;
  u.q[0] = *reinterpret_cast<const uint4*>(p);        // K kb..kb+7
  u.q[1] = *reinterpret_cast<const uint4*>(p + 16);   // K kb+16..kb+23
  return u.v;
}

// B-matrix 32x16 (KxN) sourced from row-major weight W[N,K] (so B = W^T tile):
// lane<16 -> col=lane,    K in {0..7,16..23} ; lane>=16 -> col=lane-16, K in {8..15,24..31}
__device__ __forceinline__ v16bf load_b_frag(const bf16_t* W, int ldw) {
  const int lane = threadIdx.x & 31;
  const int n  = lane & 15;
  const int kb = (lane >> 4) << 3;         // 0 or 8
  const bf16_t* p = W + n * ldw + kb;
  union { v16bf v; uint4 q[2]; } u;
  u.q[0] = *reinterpret_cast<const uint4*>(p);
  u.q[1] = *reinterpret_cast<const uint4*>(p + 16);
  return u.v;
}

// reuse-A hint as template param so the builtin always sees a constant
template <bool RA>
__device__ __forceinline__ v8f wmma_bf16(v16bf a, v16bf b, v8f c) {
  return __builtin_amdgcn_wmma_f32_16x16x32_bf16(
      /*neg_a=*/false, a, /*neg_b=*/false, b,
      /*c_mod=*/(short)0, c, /*reuse_a=*/RA, /*reuse_b=*/false);
}

// CDNA5 async global->LDS copy (16B per lane), ASYNCcnt tracked.
__device__ __forceinline__ void async_g2l_b128(void* lds, const void* g) {
  unsigned loff = (unsigned)(unsigned long long)lds;  // flat shared: low 32b = LDS addr
  asm volatile("global_load_async_to_lds_b128 %0, %1, off"
               :: "v"(loff), "v"(g) : "memory");
}
__device__ __forceinline__ void wait_async0() {
  asm volatile("s_wait_asynccnt 0x0" ::: "memory");
}

// ---------------------------------------------------------------------------
// Conversion / permutation kernels
// Gate permutation: original 4H row p (gate g = p>>9, hidden j = p&511) moves
// to np = (j>>6)*256 + g*64 + (j&63)  so each wave's 256-column strip holds
// all four gates for a contiguous 64-wide hidden range.
// ---------------------------------------------------------------------------
__global__ void cvt_f32_to_bf16(const float* __restrict__ in,
                                bf16_t* __restrict__ out, int n) {
  int i = blockIdx.x * 256 + threadIdx.x;
  if (i < n) out[i] = (bf16_t)in[i];
}

__global__ void cvt_gate_permute(const float* __restrict__ in,
                                 bf16_t* __restrict__ out, int Kdim) {
  int i = blockIdx.x * 256 + threadIdx.x;
  if (i >= LSTM_G * Kdim) return;
  int p = i / Kdim, k = i - p * Kdim;
  int g = p >> 9, j = p & 511;
  int np = ((j >> 6) << 8) | (g << 6) | (j & 63);
  out[(size_t)np * Kdim + k] = (bf16_t)in[i];
}

__global__ void bias_gate_permute(const float* __restrict__ b0,
                                  const float* __restrict__ b1,
                                  float* __restrict__ out) {
  int p = blockIdx.x * 256 + threadIdx.x;
  if (p >= LSTM_G) return;
  int g = p >> 9, j = p & 511;
  int np = ((j >> 6) << 8) | (g << 6) | (j & 63);
  out[np] = b0[p] + b1[p];
}

// ---------------------------------------------------------------------------
// WMMA GEMM:  outF[M,N] = A[M,K](bf16) * W[N,K]^T(bf16) + bias
// grid = (N/128, M/128), 256 threads = 8 waves; wave w owns rows m0..m0+15.
// The 128-col B panel (shared by all 8 waves) is staged into LDS with
// double-buffered async copies; A frags register-prefetched; B frags read
// from LDS through a rolling distance-1 window so ds loads hide behind WMMA.
// ---------------------------------------------------------------------------
__global__ __launch_bounds__(256) void wmma_gemm_bias(
    const bf16_t* __restrict__ A, int lda,
    const bf16_t* __restrict__ W, int ldw,
    const float* __restrict__ bias0, const float* __restrict__ bias1,
    float* __restrict__ outF, int ldo, int K) {
  __shared__ alignas(16) bf16_t bpan[2][128 * 32];   // 2 x 8KB
  const int tid  = threadIdx.x;
  const int wave = tid >> 5;
  const int lane = tid & 31;
  const int n0 = blockIdx.x * 128;
  const int m0 = blockIdx.y * 128 + wave * 16;
  v8f acc[8] = {};
  const bf16_t* Arow = A + (size_t)m0 * lda;

  const int sr = tid >> 2;             // staging row 0..63
  const int sc = (tid & 3) * 8;        // 16B chunk within 32-elem K slice
  auto stage = [&](int buf, int k) {
    async_g2l_b128(&bpan[buf][sr * 32 + sc],
                   W + (size_t)(n0 + sr) * ldw + k + sc);
    async_g2l_b128(&bpan[buf][(64 + sr) * 32 + sc],
                   W + (size_t)(n0 + 64 + sr) * ldw + k + sc);
  };

  stage(0, 0);
  wait_async0();
  __syncthreads();
  v16bf a_cur = load_a_frag(Arow, lda);

  int buf = 0;
  for (int k = 0; k < K; k += 32) {
    const bool more = (k + 32) < K;
    if (more) stage(buf ^ 1, k + 32);                 // async fill next buffer
    v16bf a_nxt = a_cur;
    if (more) a_nxt = load_a_frag(Arow + k + 32, lda); // reg prefetch next A

    // rolling distance-1 window over the 8 LDS B fragments
    v16bf bc = load_b_frag(&bpan[buf][0], 32);
#pragma unroll
    for (int nt = 0; nt < 8; ++nt) {
      v16bf bn = bc;
      if (nt < 7) bn = load_b_frag(&bpan[buf][(nt + 1) * 16 * 32], 32);
      if (nt == 0)
        acc[nt] = wmma_bf16<false>(a_cur, bc, acc[nt]);
      else
        acc[nt] = wmma_bf16<true>(a_cur, bc, acc[nt]);
      bc = bn;
    }

    a_cur = a_nxt;
    wait_async0();      // next buffer landed in LDS
    __syncthreads();    // everyone done reading current buffer
    buf ^= 1;
  }

  const int cn = lane & 15;
  const int mb = (lane >> 4) * 8;
#pragma unroll
  for (int nt = 0; nt < 8; ++nt) {
    int col = n0 + nt * 16 + cn;
    float bv = 0.f;
    if (bias0) bv += bias0[col];
    if (bias1) bv += bias1[col];
#pragma unroll
    for (int r = 0; r < 8; ++r) {
      outF[(size_t)(m0 + mb + r) * ldo + col] = acc[nt][r] + bv;
    }
  }
}

// ---------------------------------------------------------------------------
// Persistent LSTM recurrence. grid = B/16 blocks, 256 threads = 8 waves.
// h kept in LDS (bf16, feeds WMMA A-frags), c kept in registers.
// Weights/xp gate-permuted: wave w's 16 N-tiles are [i f g o] x 4 sub-tiles
// for hidden j in [w*64, w*64+64) -> activations are pure register math.
// B fragments stream from L2-resident Whh with a distance-2 register window.
// ---------------------------------------------------------------------------
__global__ __launch_bounds__(256) void lstm_recur(
    const float* __restrict__ xp,      // [B*T, 2048] gate-permuted, bias folded
    const bf16_t* __restrict__ Whh,    // [2048, 512] gate-permuted rows, bf16
    bf16_t* __restrict__ out_bf) {     // [B*T, 512] hidden states (bf16)
  __shared__ alignas(16) bf16_t h_s[16 * LSTM_H];    // 16 KB
  const int tid  = threadIdx.x;
  const int wave = tid >> 5;
  const int lane = tid & 31;
  const int b0 = blockIdx.x * 16;
  const int n0 = wave * 256;
  const int cn = lane & 15;
  const int mb = (lane >> 4) * 8;

  for (int i = tid; i < 16 * LSTM_H; i += 256) h_s[i] = (bf16_t)0.f;
  float creg[4][8] = {};
  __syncthreads();

  const bf16_t* Wp = Whh + (size_t)n0 * LSTM_H;

  for (int t = 0; t < LSTM_T; ++t) {
    v8f acc[16] = {};
    // rolling distance-2 B prefetch window over frag index f = (kk/32)*16+nt
    v16bf b0f = load_b_frag(Wp + 0 * 16 * LSTM_H + 0, LSTM_H);
    v16bf b1f = load_b_frag(Wp + 1 * 16 * LSTM_H + 0, LSTM_H);
    v16bf a_cur = load_a_frag(h_s, LSTM_H);
    for (int kk = 0; kk < LSTM_H; kk += 32) {
      v16bf a_nxt = a_cur;
      if (kk + 32 < LSTM_H) a_nxt = load_a_frag(h_s + kk + 32, LSTM_H);
#pragma unroll
      for (int nt = 0; nt < 16; ++nt) {
        // prefetch frag (kk,nt)+2
        int nt2 = (nt + 2) & 15;
        int kk2 = kk + (((nt + 2) >> 4) << 5);
        if (kk2 > LSTM_H - 32) kk2 = LSTM_H - 32;
        v16bf b2f = load_b_frag(Wp + (size_t)nt2 * 16 * LSTM_H + kk2, LSTM_H);
        if (nt == 0)
          acc[nt] = wmma_bf16<false>(a_cur, b0f, acc[nt]);
        else
          acc[nt] = wmma_bf16<true>(a_cur, b0f, acc[nt]);
        b0f = b1f;
        b1f = b2f;
      }
      a_cur = a_nxt;
    }
    __syncthreads();   // all h_s reads done before overwriting

#pragma unroll
    for (int u = 0; u < 4; ++u) {
      const int j = (wave << 6) | (u << 4) | cn;          // hidden index
      const int colbase = (wave << 8) | (u << 4) | cn;    // permuted gate col
#pragma unroll
      for (int r = 0; r < 8; ++r) {
        const int m = mb + r;
        const size_t xr = ((size_t)(b0 + m) * LSTM_T + t) * LSTM_G;
        float gi = acc[u     ][r] + xp[xr + colbase];
        float gf = acc[4 + u ][r] + xp[xr + colbase + 64];
        float gg = acc[8 + u ][r] + xp[xr + colbase + 128];
        float go = acc[12 + u][r] + xp[xr + colbase + 192];
        float iv = 1.f / (1.f + __expf(-gi));
        float fv = 1.f / (1.f + __expf(-gf));
        float gv = tanhf(gg);
        float ov = 1.f / (1.f + __expf(-go));
        float cc = fv * creg[u][r] + iv * gv;
        creg[u][r] = cc;
        float hh = ov * tanhf(cc);
        bf16_t hb = (bf16_t)hh;
        h_s[m * LSTM_H + j] = hb;
        out_bf[((size_t)(b0 + m) * LSTM_T + t) * LSTM_H + j] = hb;
      }
    }
    __syncthreads();   // h_s for step t complete
  }
}

// ---------------------------------------------------------------------------
// Epilogue kernels (tiny FLOP count; plain VALU)
// ---------------------------------------------------------------------------
__global__ __launch_bounds__(256) void q_last_kernel(
    const bf16_t* __restrict__ o2, const float* __restrict__ Wq,
    const float* __restrict__ bq, float* __restrict__ q) {
  const int b = blockIdx.x, tid = threadIdx.x;
  const bf16_t* h = o2 + ((size_t)b * LSTM_T + (LSTM_T - 1)) * LSTM_H;
  for (int o = tid; o < LSTM_H; o += 256) {
    float s = bq[o];
    const float* wr = Wq + (size_t)o * LSTM_H;
    for (int k = 0; k < LSTM_H; ++k) s += (float)h[k] * wr[k];
    q[b * LSTM_H + o] = s;
  }
}

__global__ __launch_bounds__(256) void attn_kernel(
    const float* __restrict__ q, const float* __restrict__ kbuf,
    const float* __restrict__ vbuf, float* __restrict__ ctx) {
  const int b = blockIdx.x, tid = threadIdx.x;
  __shared__ float p_s[LSTM_T];
  __shared__ float red[256];
  const float* qb = q + b * LSTM_H;
  const float scale = 0.044194173824159216f;  // 1/sqrt(512)
  for (int t = tid; t < LSTM_T; t += 256) {
    const float* kr = kbuf + ((size_t)b * LSTM_T + t) * LSTM_H;
    float s = 0.f;
    for (int h = 0; h < LSTM_H; ++h) s += qb[h] * kr[h];
    p_s[t] = s * scale;
  }
  __syncthreads();
  float m = -1e30f;
  for (int t = tid; t < LSTM_T; t += 256) m = fmaxf(m, p_s[t]);
  red[tid] = m; __syncthreads();
  for (int s2 = 128; s2 > 0; s2 >>= 1) {
    if (tid < s2) red[tid] = fmaxf(red[tid], red[tid + s2]);
    __syncthreads();
  }
  m = red[0]; __syncthreads();
  float sum = 0.f;
  for (int t = tid; t < LSTM_T; t += 256) {
    float e = __expf(p_s[t] - m); p_s[t] = e; sum += e;
  }
  red[tid] = sum; __syncthreads();
  for (int s2 = 128; s2 > 0; s2 >>= 1) {
    if (tid < s2) red[tid] += red[tid + s2];
    __syncthreads();
  }
  const float inv = 1.f / red[0];
  __syncthreads();
  for (int j = tid; j < LSTM_H; j += 256) {
    float a = 0.f;
    for (int t = 0; t < LSTM_T; ++t)
      a += p_s[t] * vbuf[((size_t)b * LSTM_T + t) * LSTM_H + j];
    ctx[b * LSTM_H + j] = a * inv;
  }
}

__global__ __launch_bounds__(128) void fc_kernel(
    const float* __restrict__ ctx, const float* __restrict__ Wfc,
    const float* __restrict__ bfc, float* __restrict__ out) {
  const int b = blockIdx.x, o = threadIdx.x;
  float s = bfc[o];
  const float* c = ctx + b * LSTM_H;
  const float* w = Wfc + (size_t)o * LSTM_H;
  for (int k = 0; k < LSTM_H; ++k) s += c[k] * w[k];
  out[b * LSTM_O + o] = s;
}

// ---------------------------------------------------------------------------
extern "C" void kernel_launch(void* const* d_in, const int* in_sizes, int n_in,
                              void* d_out, int out_size, void* d_ws, size_t ws_size,
                              hipStream_t stream) {
  const float* x    = (const float*)d_in[0];
  const float* Wih0 = (const float*)d_in[1];
  const float* Whh0 = (const float*)d_in[2];
  const float* bih0 = (const float*)d_in[3];
  const float* bhh0 = (const float*)d_in[4];
  const float* Wih1 = (const float*)d_in[5];
  const float* Whh1 = (const float*)d_in[6];
  const float* bih1 = (const float*)d_in[7];
  const float* bhh1 = (const float*)d_in[8];
  const float* Wq   = (const float*)d_in[9];
  const float* bq   = (const float*)d_in[10];
  const float* Wk   = (const float*)d_in[11];
  const float* bk   = (const float*)d_in[12];
  const float* Wv   = (const float*)d_in[13];
  const float* bv   = (const float*)d_in[14];
  const float* Wfc  = (const float*)d_in[15];
  const float* bfc  = (const float*)d_in[16];
  float* out = (float*)d_out;

  char* p = (char*)d_ws;
  auto carve = [&](size_t bytes) -> void* {
    void* r = (void*)p; p += (bytes + 255) & ~(size_t)255; return r;
  };
  bf16_t* wih0_b = (bf16_t*)carve((size_t)LSTM_G * LSTM_I * 2);
  bf16_t* whh0_b = (bf16_t*)carve((size_t)LSTM_G * LSTM_H * 2);
  bf16_t* wih1_b = (bf16_t*)carve((size_t)LSTM_G * LSTM_H * 2);
  bf16_t* whh1_b = (bf16_t*)carve((size_t)LSTM_G * LSTM_H * 2);
  bf16_t* wk_b   = (bf16_t*)carve((size_t)LSTM_H * LSTM_H * 2);
  bf16_t* wv_b   = (bf16_t*)carve((size_t)LSTM_H * LSTM_H * 2);
  float*  bias0p = (float*)carve((size_t)LSTM_G * 4);
  float*  bias1p = (float*)carve((size_t)LSTM_G * 4);
  bf16_t* xbf    = (bf16_t*)carve((size_t)LSTM_B * LSTM_T * LSTM_H * 2); // x, later out2
  bf16_t* obf    = (bf16_t*)carve((size_t)LSTM_B * LSTM_T * LSTM_H * 2); // layer0 hidden
  float*  xp     = (float*)carve((size_t)LSTM_B * LSTM_T * LSTM_G * 4);
  float*  kbuf   = (float*)carve((size_t)LSTM_B * LSTM_T * LSTM_H * 4);
  float*  vbuf   = (float*)carve((size_t)LSTM_B * LSTM_T * LSTM_H * 4);
  float*  qbuf   = (float*)carve((size_t)LSTM_B * LSTM_H * 4);
  float*  ctx    = (float*)carve((size_t)LSTM_B * LSTM_H * 4);

  const int NT = 256;
  const int MROWS = LSTM_B * LSTM_T;  // 32768

  cvt_f32_to_bf16<<<(MROWS * LSTM_I + NT - 1) / NT, NT, 0, stream>>>(x, xbf, MROWS * LSTM_I);
  cvt_gate_permute<<<(LSTM_G * LSTM_I + NT - 1) / NT, NT, 0, stream>>>(Wih0, wih0_b, LSTM_I);
  cvt_gate_permute<<<(LSTM_G * LSTM_H + NT - 1) / NT, NT, 0, stream>>>(Whh0, whh0_b, LSTM_H);
  cvt_gate_permute<<<(LSTM_G * LSTM_H + NT - 1) / NT, NT, 0, stream>>>(Wih1, wih1_b, LSTM_H);
  cvt_gate_permute<<<(LSTM_G * LSTM_H + NT - 1) / NT, NT, 0, stream>>>(Whh1, whh1_b, LSTM_H);
  bias_gate_permute<<<LSTM_G / NT, NT, 0, stream>>>(bih0, bhh0, bias0p);
  bias_gate_permute<<<LSTM_G / NT, NT, 0, stream>>>(bih1, bhh1, bias1p);
  cvt_f32_to_bf16<<<(LSTM_H * LSTM_H + NT - 1) / NT, NT, 0, stream>>>(Wk, wk_b, LSTM_H * LSTM_H);
  cvt_f32_to_bf16<<<(LSTM_H * LSTM_H + NT - 1) / NT, NT, 0, stream>>>(Wv, wv_b, LSTM_H * LSTM_H);

  // Layer 0: xp = x @ Wih0^T + (bih0+bhh0)   [32768 x 2048], K=128
  wmma_gemm_bias<<<dim3(LSTM_G / 128, MROWS / 128), 256, 0, stream>>>(
      xbf, LSTM_I, wih0_b, LSTM_I, bias0p, nullptr, xp, LSTM_G, LSTM_I);
  lstm_recur<<<LSTM_B / 16, 256, 0, stream>>>(xp, whh0_b, obf);

  // Layer 1: xp = out0 @ Wih1^T + (bih1+bhh1)  K=512
  wmma_gemm_bias<<<dim3(LSTM_G / 128, MROWS / 128), 256, 0, stream>>>(
      obf, LSTM_H, wih1_b, LSTM_H, bias1p, nullptr, xp, LSTM_G, LSTM_H);
  lstm_recur<<<LSTM_B / 16, 256, 0, stream>>>(xp, whh1_b, xbf);  // out2 -> xbf

  // Attention: K,V projections as WMMA GEMMs; Q only for last timestep
  wmma_gemm_bias<<<dim3(LSTM_H / 128, MROWS / 128), 256, 0, stream>>>(
      xbf, LSTM_H, wk_b, LSTM_H, bk, nullptr, kbuf, LSTM_H, LSTM_H);
  wmma_gemm_bias<<<dim3(LSTM_H / 128, MROWS / 128), 256, 0, stream>>>(
      xbf, LSTM_H, wv_b, LSTM_H, bv, nullptr, vbuf, LSTM_H, LSTM_H);
  q_last_kernel<<<LSTM_B, 256, 0, stream>>>(xbf, Wq, bq, qbuf);
  attn_kernel<<<LSTM_B, 256, 0, stream>>>(qbuf, kbuf, vbuf, ctx);
  fc_kernel<<<LSTM_B, LSTM_O, 0, stream>>>(ctx, Wfc, bfc, out);

  (void)in_sizes; (void)n_in; (void)out_size; (void)ws_size;
}
